// ByteMoE_55997783605725
// MI455X (gfx1250) — compile-verified, matrix operand order checked by reading, not run on hardware
//
#include <hip/hip_runtime.h>
#include <math.h>

typedef __bf16 bf16;
typedef __attribute__((ext_vector_type(16))) __bf16 v16bf;
typedef __attribute__((ext_vector_type(8)))  float  v8f;

#define E_EXPERTS 8
#define H_DIM     1024
#define FF_DIM    4096
#define CAP       512
#define N_TOK     8192
#define COMB_TOK  2048

// ---- CDNA5 async global->LDS copy (ASYNCcnt path, §15.18.3 op 98) ---------
__device__ __forceinline__ void async_g2l_b128(void* lds, const void* g) {
  asm volatile("global_load_async_to_lds_b128 %0, %1, off"
               :: "v"((unsigned)(unsigned long long)(uintptr_t)lds), "v"(g)
               : "memory");
}
__device__ __forceinline__ void wait_asynccnt0() {
  asm volatile("s_wait_asynccnt 0x0" ::: "memory");
}

// --------------------------------------------------------------------------
// Kernel 1: gating (softmax + stable descending sort) + dispatch into bf16
// expert buffers.  backup_k==E collapses routing: mask <=> token<512, slot==c.
// buf[e][c][:] = x[4c + r/2][:] * norm[c][r],  r = rank of expert e at token c
// --------------------------------------------------------------------------
__global__ __launch_bounds__(256) void k_gate_dispatch(
    const float* __restrict__ x, const float* __restrict__ Wg,
    const float* __restrict__ bg, bf16* __restrict__ Abuf,
    int* __restrict__ cmap)
{
  const int c   = blockIdx.x;     // active token 0..511
  const int tid = threadIdx.x;
  __shared__ float red[256][E_EXPERTS];
  __shared__ float snorm[E_EXPERTS];
  __shared__ int   sord[E_EXPERTS];

  float acc[E_EXPERTS];
#pragma unroll
  for (int e = 0; e < E_EXPERTS; ++e) acc[e] = 0.f;
  for (int h = tid; h < H_DIM; h += 256) {
    float xv = x[(size_t)c * H_DIM + h];
#pragma unroll
    for (int e = 0; e < E_EXPERTS; ++e) acc[e] += xv * Wg[h * E_EXPERTS + e];
  }
#pragma unroll
  for (int e = 0; e < E_EXPERTS; ++e) red[tid][e] = acc[e];
  __syncthreads();
  for (int s = 128; s > 0; s >>= 1) {
    if (tid < s) {
#pragma unroll
      for (int e = 0; e < E_EXPERTS; ++e) red[tid][e] += red[tid + s][e];
    }
    __syncthreads();
  }
  if (tid == 0) {
    float logit[E_EXPERTS], sc[E_EXPERTS];
    float m = -1e30f;
    for (int e = 0; e < E_EXPERTS; ++e) {
      logit[e] = red[0][e] + bg[e];
      m = fmaxf(m, logit[e]);
    }
    float ps = 0.f;
    for (int e = 0; e < E_EXPERTS; ++e) { sc[e] = expf(logit[e] - m); ps += sc[e]; }
    float tot = 0.f;
    for (int e = 0; e < E_EXPERTS; ++e) { sc[e] /= ps; tot += sc[e]; }
    bool used[E_EXPERTS] = {};
    for (int r = 0; r < E_EXPERTS; ++r) {   // stable descending (ties -> lower idx)
      int best = -1;
      for (int e = 0; e < E_EXPERTS; ++e)
        if (!used[e] && (best < 0 || sc[e] > sc[best])) best = e;
      used[best] = true;
      sord[r]  = best;
      snorm[r] = sc[best] / (tot + 1e-9f);
    }
  }
  __syncthreads();

#pragma unroll
  for (int r = 0; r < E_EXPERTS; ++r) {
    const int   e   = sord[r];
    const float sca = snorm[r];
    const float* src = x + (size_t)(4 * c + (r >> 1)) * H_DIM;  // tok = f//K bug
    bf16* dst = Abuf + ((size_t)e * CAP + c) * H_DIM;
    for (int h = tid; h < H_DIM; h += 256)
      dst[h] = (bf16)(src[h] * sca);
  }
  if (tid < 4) {                 // combine map: token 4c+j gets ranks 2j, 2j+1
    int t = 4 * c + tid;
    cmap[2 * t + 0] = sord[2 * tid + 0];
    cmap[2 * t + 1] = sord[2 * tid + 1];
  }
}

// --------------------------------------------------------------------------
// One-time weight convert + transpose: W[e][K][N] f32 -> Wt[e][N][K] bf16.
// 32x32 tiles through LDS; coalesced loads (n fast) and 16B stores (k fast).
// --------------------------------------------------------------------------
__global__ __launch_bounds__(256) void k_convert_w(
    const float* __restrict__ W, bf16* __restrict__ Wt, int K, int N)
{
  const int e  = blockIdx.z;
  const int k0 = blockIdx.y * 32;
  const int n0 = blockIdx.x * 32;
  const int tid = threadIdx.x;
  __shared__ bf16 tl[32][40];     // [n][k], padded rows keep 16B alignment
  const float* We  = W  + (size_t)e * K * N;
  bf16*        Wte = Wt + (size_t)e * N * K;
#pragma unroll
  for (int p = 0; p < 4; ++p) {
    int idx = tid + p * 256;      // 0..1023
    int i = idx >> 5, j = idx & 31;
    tl[j][i] = (bf16)We[(size_t)(k0 + i) * N + (n0 + j)];
  }
  __syncthreads();
  if (tid < 128) {
    int row = tid >> 2;           // n within tile
    int c   = (tid & 3) * 8;      // k chunk (8 bf16 = 16B)
    *(uint4*)(Wte + (size_t)(n0 + row) * K + k0 + c) = *(const uint4*)&tl[row][c];
  }
}

// --------------------------------------------------------------------------
// Grouped GEMM via v_wmma_f32_16x16x32_bf16, double-buffered async tiles.
// C[e] = act(A[e] (MxK bf16) @ Bt[e]^T (Bt is NxK bf16) + bias[e])
// 256 threads (8 waves) -> 128x64 output tile; wave w owns rows [16w,16w+16).
// Tiles for step kt+1 stream into the alternate LDS buffer via
// global_load_async_to_lds_b128 while step kt's 4 WMMAs issue; one
// s_wait_asynccnt + one barrier per K-step.  Fragments are two contiguous
// ds_load_b128 per the ISA 16-bit A/B layout (lane<16: K 0-7,16-23;
// lane>=16: K 8-15,24-31).
// --------------------------------------------------------------------------
template <bool GELU, bool OUT_BF16>
__global__ __launch_bounds__(256) void k_moe_gemm(
    const bf16*  __restrict__ A,     // [E][M][K] bf16
    const bf16*  __restrict__ Bt,    // [E][N][K] bf16 (pre-transposed)
    const float* __restrict__ bias,  // [E][N]
    void*        __restrict__ Cout,  // [E][M][N] bf16 or f32
    int M, int N, int K)
{
  const int e    = blockIdx.z;
  const int n0   = blockIdx.x * 64;
  const int m0   = blockIdx.y * 128;
  const int tid  = threadIdx.x;
  const int lane = tid & 31;
  const int wave = tid >> 5;

  __shared__ bf16 lA[2][128][32];   // [buf][m][k]
  __shared__ bf16 lB[2][64][32];    // [buf][n][k]

  const bf16* Ae = A  + (size_t)e * M * K;
  const bf16* Be = Bt + (size_t)e * N * K;

  // per-thread copy assignment (16B chunks)
  const int ar0 = (tid * 2) >> 2, ac0 = ((tid * 2) & 3) * 8;       // A chunk 0
  const int ar1 = (tid * 2 + 1) >> 2, ac1 = ((tid * 2 + 1) & 3) * 8; // A chunk 1
  const int br  = tid >> 2, bc = (tid & 3) * 8;                    // B chunk

  v8f acc[4] = {};
  const int nK = K / 32;

  // prologue: stage tile 0 into buffer 0
  {
    async_g2l_b128(&lA[0][ar0][ac0], Ae + (size_t)(m0 + ar0) * K + ac0);
    async_g2l_b128(&lA[0][ar1][ac1], Ae + (size_t)(m0 + ar1) * K + ac1);
    async_g2l_b128(&lB[0][br][bc],   Be + (size_t)(n0 + br) * K + bc);
  }
  wait_asynccnt0();
  __syncthreads();

  for (int kt = 0; kt < nK; ++kt) {
    const int cur = kt & 1;
    // stream tiles for kt+1 into the other buffer (overlaps with WMMAs below)
    if (kt + 1 < nK) {
      const int k1 = (kt + 1) * 32;
      async_g2l_b128(&lA[cur ^ 1][ar0][ac0], Ae + (size_t)(m0 + ar0) * K + k1 + ac0);
      async_g2l_b128(&lA[cur ^ 1][ar1][ac1], Ae + (size_t)(m0 + ar1) * K + k1 + ac1);
      async_g2l_b128(&lB[cur ^ 1][br][bc],   Be + (size_t)(n0 + br) * K + k1 + bc);
      if (kt + 2 < nK)   // keep L2 ahead of the async engine
        __builtin_prefetch(Be + (size_t)(n0 + br) * K + (kt + 2) * 32 + bc, 0, 1);
    }

    // compute from current buffer
    const int arow = wave * 16 + (lane & 15);
    const int kb   = (lane >= 16) ? 8 : 0;
    union { v16bf v; uint4 q[2]; } ua;
    ua.q[0] = *(const uint4*)&lA[cur][arow][kb];
    ua.q[1] = *(const uint4*)&lA[cur][arow][kb + 16];
#pragma unroll
    for (int t = 0; t < 4; ++t) {
      const int bcol = t * 16 + (lane & 15);
      union { v16bf v; uint4 q[2]; } ub;
      ub.q[0] = *(const uint4*)&lB[cur][bcol][kb];
      ub.q[1] = *(const uint4*)&lB[cur][bcol][kb + 16];
      acc[t] = __builtin_amdgcn_wmma_f32_16x16x32_bf16(
          false, ua.v, false, ub.v, (short)0, acc[t], false, false);
    }

    wait_asynccnt0();   // next buffer resident
    __syncthreads();    // all waves done reading cur; safe to overwrite next iter
  }

  // Epilogue: C/D layout -> element v: row m0+16w+v (+8 for upper half-lanes)
#pragma unroll
  for (int t = 0; t < 4; ++t) {
    const int   colg = n0 + t * 16 + (lane & 15);
    const float bv   = bias[(size_t)e * N + colg];
#pragma unroll
    for (int v = 0; v < 8; ++v) {
      int   rowg = m0 + wave * 16 + v + ((lane >= 16) ? 8 : 0);
      float val  = acc[t][v] + bv;
      if (GELU)
        val = 0.5f * val * (1.0f + erff(val * 0.70710678118654752f));
      size_t idx = ((size_t)e * M + rowg) * N + colg;
      if (OUT_BF16) ((bf16*)Cout)[idx]  = (bf16)val;
      else          ((float*)Cout)[idx] = val;
    }
  }
}

// --------------------------------------------------------------------------
// Combine: y[t] = out[e0][t/4] + out[e1][t/4] for t < 2048, zero elsewhere.
// Aux loss is exactly 0 (em is all-ones since backup_k == E).
// --------------------------------------------------------------------------
__global__ __launch_bounds__(256) void k_combine(
    const float* __restrict__ Outb, const int* __restrict__ cmap,
    float* __restrict__ y)
{
  const int t   = blockIdx.x;
  const int tid = threadIdx.x;
  if (t < COMB_TOK) {
    const int c  = t >> 2;
    const int e0 = cmap[2 * t + 0];
    const int e1 = cmap[2 * t + 1];
    const float* o0 = Outb + ((size_t)e0 * CAP + c) * H_DIM;
    const float* o1 = Outb + ((size_t)e1 * CAP + c) * H_DIM;
    for (int h = tid; h < H_DIM; h += 256)
      y[(size_t)t * H_DIM + h] = o0[h] + o1[h];
  } else {
    for (int h = tid; h < H_DIM; h += 256)
      y[(size_t)t * H_DIM + h] = 0.f;
  }
  if (t == 0 && tid == 0)
    y[(size_t)N_TOK * H_DIM] = 0.0f;   // aux loss scalar
}

// --------------------------------------------------------------------------
extern "C" void kernel_launch(void* const* d_in, const int* in_sizes, int n_in,
                              void* d_out, int out_size, void* d_ws, size_t ws_size,
                              hipStream_t stream) {
  const float* x  = (const float*)d_in[0];
  const float* Wg = (const float*)d_in[1];
  const float* bg = (const float*)d_in[2];
  const float* W1 = (const float*)d_in[3];
  const float* b1 = (const float*)d_in[4];
  const float* W2 = (const float*)d_in[5];
  const float* b2 = (const float*)d_in[6];
  float* y = (float*)d_out;

  char* ws = (char*)d_ws;
  bf16*  Abuf = (bf16*)(ws);                            //   8 MB [E][512][1024] bf16
  bf16*  Hmid = (bf16*)(ws + ((size_t)8   << 20));      //  32 MB [E][512][4096] bf16
  float* Outb = (float*)(ws + ((size_t)40 << 20));      //  16 MB [E][512][1024] f32
  int*   cmap = (int*)  (ws + ((size_t)56 << 20));      //  16 KB [2048][2]
  bf16*  W1t  = (bf16*)(ws + ((size_t)57  << 20));      //  64 MB [E][4096][1024] bf16
  bf16*  W2t  = (bf16*)(ws + ((size_t)121 << 20));      //  64 MB [E][1024][4096] bf16

  // one-time weight convert+transpose to bf16 [N][K]
  dim3 gc1(FF_DIM / 32, H_DIM / 32, E_EXPERTS);
  k_convert_w<<<gc1, 256, 0, stream>>>(W1, W1t, H_DIM, FF_DIM);
  dim3 gc2(H_DIM / 32, FF_DIM / 32, E_EXPERTS);
  k_convert_w<<<gc2, 256, 0, stream>>>(W2, W2t, FF_DIM, H_DIM);

  k_gate_dispatch<<<CAP, 256, 0, stream>>>(x, Wg, bg, Abuf, cmap);

  dim3 g1(FF_DIM / 64, CAP / 128, E_EXPERTS);           // (64, 4, 8)
  k_moe_gemm<true,  true ><<<g1, 256, 0, stream>>>(Abuf, W1t, b1, (void*)Hmid,
                                                   CAP, FF_DIM, H_DIM);
  dim3 g2(H_DIM / 64, CAP / 128, E_EXPERTS);            // (16, 4, 8)
  k_moe_gemm<false, false><<<g2, 256, 0, stream>>>(Hmid, W2t, b2, (void*)Outb,
                                                   CAP, H_DIM, FF_DIM);

  k_combine<<<N_TOK, 256, 0, stream>>>(Outb, cmap, y);
  (void)in_sizes; (void)n_in; (void)out_size; (void)ws_size;
}
